// Qwen2VLVisionBlock_36412732736110
// MI455X (gfx1250) — compile-verified, matrix operand order checked by reading, not run on hardware
//
#include <hip/hip_runtime.h>

// ---------------------------------------------------------------------------
// Qwen2VL vision block for MI455X (gfx1250): all GEMMs + attention through
// v_wmma_f32_16x16x32_bf16, fp32 accumulation, flash-style segment attention.
// GEMM tiles double-buffered in LDS and filled with GLOBAL_LOAD_ASYNC_TO_LDS
// (ASYNCcnt) when the toolchain exposes the builtins.
// ---------------------------------------------------------------------------

#define DIM        1280
#define HEADS      16
#define HEAD_DIM   80
#define HDP        96          // head dim padded to 3*32 for WMMA K
#define MLP_HIDDEN 5120
#define SEQ        3072
#define QKV_N      (3 * DIM)   // 3840

typedef __attribute__((ext_vector_type(16))) __bf16 bf16x16;
typedef __attribute__((ext_vector_type(8)))  __bf16 bf16x8;
typedef __attribute__((ext_vector_type(8)))  float  f32x8;

// ---- async global->LDS path (gfx1250) -------------------------------------
#if defined(__gfx1250__) && __has_builtin(__builtin_amdgcn_global_load_async_to_lds_b128)
#define HAVE_ASYNC_LDS 1
typedef int v4i_t __attribute__((__vector_size__(16)));
typedef __attribute__((address_space(1))) v4i_t g_v4i_t;
typedef __attribute__((address_space(3))) v4i_t l_v4i_t;
__device__ __forceinline__ void async_copy16(__bf16* lds_dst, const __bf16* gsrc) {
    __builtin_amdgcn_global_load_async_to_lds_b128(
        (g_v4i_t*)gsrc, (l_v4i_t*)lds_dst, 0, 0);
}
template <int N>
__device__ __forceinline__ void wait_async() {
#if __has_builtin(__builtin_amdgcn_s_wait_asynccnt)
    __builtin_amdgcn_s_wait_asynccnt(N);
#else
    asm volatile("s_wait_asynccnt %0" :: "i"(N) : "memory");
#endif
}
#else
#define HAVE_ASYNC_LDS 0
__device__ __forceinline__ void async_copy16(__bf16* lds_dst, const __bf16* gsrc) {
    *(uint4*)lds_dst = *(const uint4*)gsrc;      // sync fallback
}
template <int N>
__device__ __forceinline__ void wait_async() {}
#endif

// Load one 16x16x32 bf16 A/B fragment from an LDS tile (row-major, `stride`
// elements per row).  ISA 16-bit A layout: lanes 0-15 rows, elements [0..7]
// are K = (lane<16?0:8)+0..7, elements [8..15] are the same +16.
__device__ __forceinline__ bf16x16 load_frag(const __bf16* base, int lane, int stride) {
    int row = lane & 15;
    int kg  = (lane >> 4) << 3;          // 0 or 8
    const __bf16* p = base + row * stride + kg;
    bf16x8 lo = *(const bf16x8*)(p);
    bf16x8 hi = *(const bf16x8*)(p + 16);
    bf16x16 r;
#pragma unroll
    for (int i = 0; i < 8; ++i) { r[i] = lo[i]; r[8 + i] = hi[i]; }
    return r;
}

__device__ __forceinline__ f32x8 wmma_bf16(bf16x16 a, bf16x16 b, f32x8 c) {
    return __builtin_amdgcn_wmma_f32_16x16x32_bf16(false, a, false, b,
                                                   (short)0, c, false, false);
}

// ---------------------------------------------------------------------------
// Weight convert + transpose: W[K][N] fp32 -> Wt[N][K] bf16.
// Grid (K/256, N): coalesced writes, no integer div/mod.
// ---------------------------------------------------------------------------
__global__ void convert_transpose_kernel(const float* __restrict__ W,
                                         __bf16* __restrict__ Wt,
                                         int K, int N) {
    int k = blockIdx.x * 256 + threadIdx.x;
    int n = blockIdx.y;
    Wt[(size_t)n * K + k] = (__bf16)W[(size_t)k * N + n];
}

// ---------------------------------------------------------------------------
// LayerNorm: one 256-thread block per row, bf16 output for WMMA A-matrix.
// ---------------------------------------------------------------------------
__global__ __launch_bounds__(256)
void layernorm_kernel(const float* __restrict__ x,
                      const float* __restrict__ g,
                      const float* __restrict__ b,
                      __bf16* __restrict__ out, int D) {
    int row = blockIdx.x;
    int t   = threadIdx.x;
    int w   = t >> 5, l = t & 31;
    __shared__ float red0[8], red1[8], bc[2];

    const float* xr = x + (size_t)row * D;
    float s = 0.f, s2 = 0.f;
    for (int i = t; i < D; i += 256) { float v = xr[i]; s += v; s2 += v * v; }
#pragma unroll
    for (int off = 16; off > 0; off >>= 1) {
        s  += __shfl_xor(s,  off, 32);
        s2 += __shfl_xor(s2, off, 32);
    }
    if (l == 0) { red0[w] = s; red1[w] = s2; }
    __syncthreads();
    if (t == 0) {
        float a = 0.f, c = 0.f;
        for (int i = 0; i < 8; ++i) { a += red0[i]; c += red1[i]; }
        float mean = a / D;
        float var  = c / D - mean * mean;
        bc[0] = mean;
        bc[1] = rsqrtf(var + 1e-6f);
    }
    __syncthreads();
    float mean = bc[0], inv = bc[1];
    __bf16* orow = out + (size_t)row * D;
    for (int i = t; i < D; i += 256)
        orow[i] = (__bf16)((xr[i] - mean) * inv * g[i] + b[i]);
}

// ---------------------------------------------------------------------------
// Generic WMMA GEMM:  C[M][N] = epi(A[M][K] @ Bt[N][K]^T + bias)  (+ residual)
//   EPI 0: fp32 store      EPI 1: residual add, fp32 store
//   EPI 2: quickGELU, bf16 store
// Tiles: BM=128 BN=128 BK=32; 8 waves as 4(M)x2(N); wave = 2x4 16x16 tiles.
// Double-buffered LDS, async global->LDS fills (4 b128 per thread per tile).
// ---------------------------------------------------------------------------
template <int EPI>
__global__ __launch_bounds__(256)
void gemm_bf16_kernel(const __bf16* __restrict__ A,
                      const __bf16* __restrict__ Bt,
                      const float*  __restrict__ bias,
                      const float*  __restrict__ resid,
                      float*        __restrict__ Cf,
                      __bf16*       __restrict__ Cb,
                      int M, int N, int K) {
    __shared__ __align__(16) __bf16 aT[2][128 * 32];
    __shared__ __align__(16) __bf16 bT[2][128 * 32];

    const int bm = blockIdx.y * 128;
    const int bn = blockIdx.x * 128;
    const int t  = threadIdx.x;
    const int w  = t >> 5, l = t & 31;
    const int wm = (w & 3) * 32;        // wave M offset within block tile
    const int wn = (w >> 2) * 64;       // wave N offset within block tile

    // per-thread tile-fill coordinates: 512 chunks of 8 bf16, 2 per thread
    const int r0 = t >> 2,          kc0 = (t & 3) << 3;
    const int r1 = (t + 256) >> 2,  kc1 = kc0;

    const __bf16* Arow0 = A  + (size_t)(bm + r0) * K + kc0;
    const __bf16* Arow1 = A  + (size_t)(bm + r1) * K + kc1;
    const __bf16* Brow0 = Bt + (size_t)(bn + r0) * K + kc0;
    const __bf16* Brow1 = Bt + (size_t)(bn + r1) * K + kc1;

    f32x8 acc[2][4] = {};
    const int nk = K >> 5;

    auto issue = [&](int buf, int k0) {
        async_copy16(&aT[buf][r0 * 32 + kc0], Arow0 + k0);
        async_copy16(&aT[buf][r1 * 32 + kc1], Arow1 + k0);
        async_copy16(&bT[buf][r0 * 32 + kc0], Brow0 + k0);
        async_copy16(&bT[buf][r1 * 32 + kc1], Brow1 + k0);
#if !HAVE_ASYNC_LDS
        if (k0 + 32 < K) {
            __builtin_prefetch(Arow0 + k0 + 32, 0, 1);
            __builtin_prefetch(Brow0 + k0 + 32, 0, 1);
        }
#endif
    };

    issue(0, 0);
    for (int i = 0; i < nk; ++i) {
        const int buf = i & 1;
        if (i + 1 < nk) { issue(buf ^ 1, (i + 1) << 5); wait_async<4>(); }
        else            { wait_async<0>(); }
        __syncthreads();

        bf16x16 af[2], bf_[4];
#pragma unroll
        for (int mt = 0; mt < 2; ++mt)
            af[mt] = load_frag(&aT[buf][(wm + mt * 16) * 32], l, 32);
#pragma unroll
        for (int nt = 0; nt < 4; ++nt)
            bf_[nt] = load_frag(&bT[buf][(wn + nt * 16) * 32], l, 32);
#pragma unroll
        for (int mt = 0; mt < 2; ++mt)
#pragma unroll
            for (int nt = 0; nt < 4; ++nt)
                acc[mt][nt] = wmma_bf16(af[mt], bf_[nt], acc[mt][nt]);
        __syncthreads();   // protect buf before it is refilled at i+1
    }

    // epilogue: lane l, VGPR r holds (m = r + (l>=16)*8, n = l&15)
#pragma unroll
    for (int mt = 0; mt < 2; ++mt) {
#pragma unroll
        for (int nt = 0; nt < 4; ++nt) {
            int col = bn + wn + nt * 16 + (l & 15);
            float bi = bias[col];
#pragma unroll
            for (int r = 0; r < 8; ++r) {
                int row = bm + wm + mt * 16 + r + ((l >> 4) << 3);
                float v = acc[mt][nt][r] + bi;
                if (EPI == 1) v += resid[(size_t)row * N + col];
                if (EPI == 2) v = v / (1.f + __expf(-1.702f * v));
                if (EPI == 2) Cb[(size_t)row * N + col] = (__bf16)v;
                else          Cf[(size_t)row * N + col] = v;
            }
        }
    }
}

// ---------------------------------------------------------------------------
// RoPE + pack to bf16, head-dim padded 80 -> 96 with zeros.
// qkv fp32 layout: [S][3][HEADS][80].  Grid (SEQ, HEADS), 96 threads (t = d).
// ---------------------------------------------------------------------------
__global__ __launch_bounds__(96)
void rope_pack_kernel(const float* __restrict__ qkv,
                      const float* __restrict__ rot,
                      __bf16* __restrict__ qb,
                      __bf16* __restrict__ kb,
                      __bf16* __restrict__ vb) {
    const int s = blockIdx.x;
    const int h = blockIdx.y;
    const int d = threadIdx.x;          // 0..95

    float qv = 0.f, kv = 0.f, vv = 0.f;
    if (d < HEAD_DIM) {
        const float* qp = qkv + (size_t)s * QKV_N + h * HEAD_DIM;
        const float* kp = qp + DIM;
        const float* vp = qp + 2 * DIM;
        int dm = (d < HEAD_DIM / 2) ? d : d - HEAD_DIM / 2;
        float fr = rot[(size_t)s * (HEAD_DIM / 2) + dm];
        float cs = __cosf(fr), sn = __sinf(fr);
        float t0 = qp[d];
        float rh = (d < HEAD_DIM / 2) ? -qp[d + HEAD_DIM / 2] : qp[d - HEAD_DIM / 2];
        qv = t0 * cs + rh * sn;
        t0 = kp[d];
        rh = (d < HEAD_DIM / 2) ? -kp[d + HEAD_DIM / 2] : kp[d - HEAD_DIM / 2];
        kv = t0 * cs + rh * sn;
        vv = vp[d];
    }
    size_t o = ((size_t)s * HEADS + h) * HDP + d;
    qb[o] = (__bf16)qv;
    kb[o] = (__bf16)kv;
    vb[o] = (__bf16)vv;
}

// ---------------------------------------------------------------------------
// Flash attention over block-diagonal segments.
// Grid: (SEQ/64, HEADS).  Block: 256 threads (8 waves = 4M x 2N).
// Q tile 64 rows; KV chunks of 64; K dim for scores = 96 (3 WMMA steps).
// ---------------------------------------------------------------------------
__global__ __launch_bounds__(256)
void attention_kernel(const __bf16* __restrict__ qb,
                      const __bf16* __restrict__ kb,
                      const __bf16* __restrict__ vb,
                      const int* __restrict__ cu,
                      __bf16* __restrict__ attnout) {
    const int BQ = 64, BK = 64;
    __shared__ __align__(16) __bf16 qlds[BQ * HDP];      // 12 KB
    __shared__ __align__(16) __bf16 klds[BK * HDP];      // 12 KB
    __shared__ __align__(16) __bf16 vlds[HDP * BK];      // 12 KB (transposed)
    __shared__ __align__(16) float  slds[BQ * BK];       // 16 KB
    __shared__ __align__(16) __bf16 plds[BQ * BK];       //  8 KB
    __shared__ float m_s[BQ], l_s[BQ], corr_s[BQ];

    const int head = blockIdx.y;
    const int q0   = blockIdx.x * BQ;
    const int t    = threadIdx.x;
    const int w    = t >> 5, l = t & 31;
    const int wm   = w & 3;             // M-tile (16 rows)
    const int wn   = w >> 2;            // N half

    int seg_start = 0, seg_end = SEQ;
    for (int i = 0; i < 4; ++i)
        if (q0 >= cu[i] && q0 < cu[i + 1]) { seg_start = cu[i]; seg_end = cu[i + 1]; }

    // tile-fill coords: row = t>>2, quarter covers kc = (t&3)*8 + j*32
    const int fr_ = t >> 2;
    const int fq  = (t & 3) << 3;

    // load Q tile (64 x 96)
#pragma unroll
    for (int j = 0; j < 3; ++j) {
        int kc = fq + j * 32;
        *(uint4*)&qlds[fr_ * HDP + kc] =
            *(const uint4*)&qb[((size_t)(q0 + fr_) * HEADS + head) * HDP + kc];
    }
    if (t < BQ) { m_s[t] = -1e30f; l_s[t] = 0.f; }

    f32x8 acc_o[3] = {};
    const float scale = 0.11180339887498949f;   // 1/sqrt(80)
    __syncthreads();

    for (int kv0 = seg_start; kv0 < seg_end; kv0 += BK) {
        int kvlen = seg_end - kv0;
        if (kvlen > BK) kvlen = BK;

        // K tile (zero-padded past segment end)
#pragma unroll
        for (int j = 0; j < 3; ++j) {
            int kc = fq + j * 32;
            uint4 val = make_uint4(0u, 0u, 0u, 0u);
            if (fr_ < kvlen)
                val = *(const uint4*)&kb[((size_t)(kv0 + fr_) * HEADS + head) * HDP + kc];
            *(uint4*)&klds[fr_ * HDP + kc] = val;
        }
        // V tile, transposed: vlds[d][key]; waves over rows, lanes over dims
        for (int rr = w; rr < BK; rr += 8) {
            const __bf16* vrow = vb + ((size_t)(kv0 + rr) * HEADS + head) * HDP;
#pragma unroll
            for (int d = 0; d < 3; ++d) {
                int dd = l + d * 32;
                vlds[dd * BK + rr] = (rr < kvlen) ? vrow[dd] : (__bf16)0.f;
            }
        }
        __syncthreads();

        // ---- scores: wave covers M-tile wm, N-tiles wn*2 .. wn*2+1 ----
        f32x8 acc_s[2] = {};
#pragma unroll
        for (int ks = 0; ks < 3; ++ks) {
            bf16x16 af = load_frag(&qlds[(wm * 16) * HDP + ks * 32], l, HDP);
#pragma unroll
            for (int nt = 0; nt < 2; ++nt) {
                bf16x16 bf_ = load_frag(&klds[((wn * 2 + nt) * 16) * HDP + ks * 32], l, HDP);
                acc_s[nt] = wmma_bf16(af, bf_, acc_s[nt]);
            }
        }
#pragma unroll
        for (int nt = 0; nt < 2; ++nt) {
            int nn = (wn * 2 + nt) * 16 + (l & 15);
#pragma unroll
            for (int r = 0; r < 8; ++r) {
                int mm = wm * 16 + r + ((l >> 4) << 3);
                slds[mm * BK + nn] = acc_s[nt][r] * scale;
            }
        }
        __syncthreads();

        // ---- online softmax: one thread per row ----
        if (t < BQ) {
            int row = t;
            float mo = m_s[row];
            float rmax = -1e30f;
            for (int c = 0; c < kvlen; ++c)
                rmax = fmaxf(rmax, slds[row * BK + c]);
            float mn   = fmaxf(mo, rmax);
            float corr = __expf(mo - mn);
            float sum  = 0.f;
            for (int c = 0; c < BK; ++c) {
                float p = (c < kvlen) ? __expf(slds[row * BK + c] - mn) : 0.f;
                sum += p;
                plds[row * BK + c] = (__bf16)p;
            }
            l_s[row]    = l_s[row] * corr + sum;
            m_s[row]    = mn;
            corr_s[row] = corr;
        }
        __syncthreads();

        // ---- O update: rescale then accumulate P @ V ----
#pragma unroll
        for (int nt = 0; nt < 3; ++nt)
#pragma unroll
            for (int r = 0; r < 8; ++r) {
                int mm = wm * 16 + r + ((l >> 4) << 3);
                acc_o[nt][r] *= corr_s[mm];
            }
#pragma unroll
        for (int ks = 0; ks < 2; ++ks) {
            bf16x16 af = load_frag(&plds[(wm * 16) * BK + ks * 32], l, BK);
#pragma unroll
            for (int nt = 0; nt < 3; ++nt) {
                bf16x16 bf_ = load_frag(&vlds[((wn * 3 + nt) * 16) * BK + ks * 32], l, BK);
                acc_o[nt] = wmma_bf16(af, bf_, acc_o[nt]);
            }
        }
        __syncthreads();
    }

    // ---- finalize: O / l, write bf16 [S][DIM] for proj GEMM ----
#pragma unroll
    for (int nt = 0; nt < 3; ++nt) {
        int nn = (wn * 3 + nt) * 16 + (l & 15);
        if (nn < HEAD_DIM) {
#pragma unroll
            for (int r = 0; r < 8; ++r) {
                int mm = wm * 16 + r + ((l >> 4) << 3);
                float v = acc_o[nt][r] / (l_s[mm] + 1e-30f);
                attnout[(size_t)(q0 + mm) * DIM + head * HEAD_DIM + nn] = (__bf16)v;
            }
        }
    }
}

// ---------------------------------------------------------------------------
// Host-side orchestration
// ---------------------------------------------------------------------------
extern "C" void kernel_launch(void* const* d_in, const int* in_sizes, int n_in,
                              void* d_out, int out_size, void* d_ws, size_t ws_size,
                              hipStream_t stream) {
    const float* hidden = (const float*)d_in[0];
    const int*   cu     = (const int*)d_in[1];
    const float* rot    = (const float*)d_in[2];
    const float* qkv_w  = (const float*)d_in[3];
    const float* qkv_b  = (const float*)d_in[4];
    const float* proj_w = (const float*)d_in[5];
    const float* proj_b = (const float*)d_in[6];
    const float* ln1_g  = (const float*)d_in[7];
    const float* ln1_b  = (const float*)d_in[8];
    const float* ln2_g  = (const float*)d_in[9];
    const float* ln2_b  = (const float*)d_in[10];
    const float* fc1_w  = (const float*)d_in[11];
    const float* fc1_b  = (const float*)d_in[12];
    const float* fc2_w  = (const float*)d_in[13];
    const float* fc2_b  = (const float*)d_in[14];
    float* out = (float*)d_out;

    char* ws = (char*)d_ws;
    auto carve = [&](size_t bytes) -> void* {
        void* p = (void*)ws;
        ws += (bytes + 255) & ~(size_t)255;
        return p;
    };
    __bf16* qkv_wt  = (__bf16*)carve((size_t)QKV_N * DIM * 2);
    __bf16* proj_wt = (__bf16*)carve((size_t)DIM * DIM * 2);
    __bf16* fc1_wt  = (__bf16*)carve((size_t)MLP_HIDDEN * DIM * 2);
    __bf16* fc2_wt  = (__bf16*)carve((size_t)DIM * MLP_HIDDEN * 2);
    __bf16* x1b     = (__bf16*)carve((size_t)SEQ * DIM * 2);
    float*  qkvf    = (float*) carve((size_t)SEQ * QKV_N * 4);
    __bf16* qbuf    = (__bf16*)carve((size_t)SEQ * HEADS * HDP * 2);
    __bf16* kbuf    = (__bf16*)carve((size_t)SEQ * HEADS * HDP * 2);
    __bf16* vbuf    = (__bf16*)carve((size_t)SEQ * HEADS * HDP * 2);
    __bf16* attnb   = (__bf16*)carve((size_t)SEQ * DIM * 2);
    float*  hbuf    = (float*) carve((size_t)SEQ * DIM * 4);
    __bf16* y2b     = (__bf16*)carve((size_t)SEQ * DIM * 2);
    __bf16* mlp1b   = (__bf16*)carve((size_t)SEQ * MLP_HIDDEN * 2);

    // 1) weight convert + transpose (fp32 -> bf16, [K][N] -> [N][K])
    convert_transpose_kernel<<<dim3(DIM / 256, QKV_N), 256, 0, stream>>>(qkv_w, qkv_wt, DIM, QKV_N);
    convert_transpose_kernel<<<dim3(DIM / 256, DIM), 256, 0, stream>>>(proj_w, proj_wt, DIM, DIM);
    convert_transpose_kernel<<<dim3(DIM / 256, MLP_HIDDEN), 256, 0, stream>>>(fc1_w, fc1_wt, DIM, MLP_HIDDEN);
    convert_transpose_kernel<<<dim3(MLP_HIDDEN / 256, DIM), 256, 0, stream>>>(fc2_w, fc2_wt, MLP_HIDDEN, DIM);

    // 2) LN1
    layernorm_kernel<<<SEQ, 256, 0, stream>>>(hidden, ln1_g, ln1_b, x1b, DIM);

    // 3) QKV GEMM (fp32 out for RoPE)
    gemm_bf16_kernel<0><<<dim3(QKV_N / 128, SEQ / 128), 256, 0, stream>>>(
        x1b, qkv_wt, qkv_b, nullptr, qkvf, nullptr, SEQ, QKV_N, DIM);

    // 4) RoPE + pack bf16 (pad head dim to 96)
    rope_pack_kernel<<<dim3(SEQ, HEADS), 96, 0, stream>>>(qkvf, rot, qbuf, kbuf, vbuf);

    // 5) flash attention over segments
    attention_kernel<<<dim3(SEQ / 64, HEADS), 256, 0, stream>>>(qbuf, kbuf, vbuf, cu, attnb);

    // 6) proj GEMM + residual(hidden) -> h
    gemm_bf16_kernel<1><<<dim3(DIM / 128, SEQ / 128), 256, 0, stream>>>(
        attnb, proj_wt, proj_b, hidden, hbuf, nullptr, SEQ, DIM, DIM);

    // 7) LN2
    layernorm_kernel<<<SEQ, 256, 0, stream>>>(hbuf, ln2_g, ln2_b, y2b, DIM);

    // 8) fc1 GEMM + quickGELU -> bf16
    gemm_bf16_kernel<2><<<dim3(MLP_HIDDEN / 128, SEQ / 128), 256, 0, stream>>>(
        y2b, fc1_wt, fc1_b, nullptr, nullptr, mlp1b, SEQ, MLP_HIDDEN, DIM);

    // 9) fc2 GEMM + residual(h) -> out
    gemm_bf16_kernel<1><<<dim3(DIM / 128, SEQ / 128), 256, 0, stream>>>(
        mlp1b, fc2_wt, fc2_b, hbuf, out, nullptr, SEQ, DIM, MLP_HIDDEN);
}
